// ODE_RNN_Rubanova_70179765616912
// MI455X (gfx1250) — matrix operand order repaired and checked by reading, not compiled, not run on hardware
//
#include <hip/hip_runtime.h>

// ---------------------------------------------------------------------------
// ODE-RNN (Rubanova) for MI455X / gfx1250, wave32 + WMMA f32_16x16x32_f16.
//
// Batch rows are independent through the whole recurrence, so the sequential
// chain is split into 16 INDEPENDENT workgroups (one 16-row m-tile each),
// no inter-WG sync. Per WG: 512 thr = 16 wave32, wave w owns n-tile w.
//   - W_f column block cached in VGPRs (64 regs) for the entire kernel
//   - z (fp16) in an 8.4KB LDS buffer; h/c/RK4-sum in VGPRs (8 each/lane)
//   - v_tanh_f32 / exp2 / rcp branchless epilogues
//   - async LDS->global writeback of h_t, prefetch of x[t+1]
// readout_kernel: 1024 blocks x 256 thr; fused relu(Y W_l^T + b_l) W_o^T.
// ---------------------------------------------------------------------------

typedef __attribute__((ext_vector_type(16))) _Float16 v16h;
typedef __attribute__((ext_vector_type(8)))  _Float16 v8h;
typedef __attribute__((ext_vector_type(8)))  float    v8f;

union V16 { v16h v; v8h h[2]; };

#define Hd  256
#define Bd  256
#define Fd  64
#define Cd  32
#define Td  512
#define KT  8      // K-steps for K=256 with K=32 per WMMA
#define ZS  264    // padded LDS stride (halves): 264*2 = 528 = 33*16B
#define XS  72     // padded LDS stride for 64-wide x rows: 144 = 9*16B

// ---- workspace layout (bytes) ----
#define WF16_OFF   0u
#define WIH16_OFF  (131072u)
#define WHH16_OFF  (WIH16_OFF + 131072u)
#define WL16_OFF   (WHH16_OFF + 524288u)
#define WO16_OFF   (WL16_OFF + 131072u)
#define BCOMB_OFF  (WO16_OFF + 16384u)
#define Y16_OFF    (1048576u)
#define WS_NEEDED  (Y16_OFF + (size_t)Td * Bd * Hd * 2u)

// ---------------- fast branchless activations ----------------
__device__ __forceinline__ float fast_exp2(float x) {
#if __has_builtin(__builtin_amdgcn_exp2f)
  return __builtin_amdgcn_exp2f(x);
#else
  return exp2f(x);
#endif
}
__device__ __forceinline__ float fast_rcp(float x) {
#if __has_builtin(__builtin_amdgcn_rcpf)
  return __builtin_amdgcn_rcpf(x);
#else
  return 1.0f / x;
#endif
}
__device__ __forceinline__ float fast_tanh(float x) {
#if __has_builtin(__builtin_amdgcn_tanhf)
  return __builtin_amdgcn_tanhf(x);          // gfx1250 v_tanh_f32 (confirmed)
#elif __has_builtin(__builtin_amdgcn_tanh_f32)
  return __builtin_amdgcn_tanh_f32(x);
#else
  const float ax = __builtin_fabsf(x);
  const float t  = fast_exp2(-2.885390082f * ax);
  const float r  = (1.0f - t) * fast_rcp(1.0f + t);
  return __builtin_copysignf(r, x);
#endif
}
__device__ __forceinline__ float fast_sigmoid(float x) {
  return fast_rcp(1.0f + fast_exp2(-1.44269504089f * x));
}

// A fragment: 16x32 (MxK) fp16. Lanes 0-15: M=lane, K={ks*32+0..7, +16..23};
// lanes 16-31: M=lane-16, K={ks*32+8..15, +24..31}.
__device__ __forceinline__ v16h load_a(const _Float16* buf, int stride, int rowbase,
                                       int kstep, int lane) {
  const int r  = rowbase + (lane & 15);
  const int kb = kstep * 32 + ((lane >> 4) << 3);
  const _Float16* p = buf + r * stride + kb;
  V16 a;
  a.h[0] = *(const v8h*)(p);
  a.h[1] = *(const v8h*)(p + 16);
  return a.v;
}

// B fragment: 32x16 (KxN) fp16 from W stored [n][k] row-major (z W^T).
__device__ __forceinline__ v16h load_b(const _Float16* buf, int stride, int ncolbase,
                                       int kstep, int lane) {
  const int n  = ncolbase + (lane & 15);
  const int kb = kstep * 32 + ((lane >> 4) << 3);
  const _Float16* p = buf + n * stride + kb;
  V16 b;
  b.h[0] = *(const v8h*)(p);
  b.h[1] = *(const v8h*)(p + 16);
  return b.v;
}

__device__ __forceinline__ v8f wmma_f16(v16h a, v16h b, v8f c) {
  return __builtin_amdgcn_wmma_f32_16x16x32_f16(false, a, false, b, (short)0, c,
                                                false, false);
}

// async LDS -> global store of one 16B chunk (ASYNCcnt tracked)
__device__ __forceinline__ void async_st_b128(const _Float16* gptr, const _Float16* lptr) {
  const unsigned long long ga = (unsigned long long)(uintptr_t)gptr;
  const unsigned la = (unsigned)(uintptr_t)lptr;   // low 32 bits = LDS byte address
  asm volatile("global_store_async_from_lds_b128 %0, %1, off"
               :: "v"(ga), "v"(la) : "memory");
}
__device__ __forceinline__ void wait_asynccnt0() {
  asm volatile("s_wait_asynccnt 0x0" ::: "memory");
}

// ---------------------------------------------------------------------------
__global__ void prep_kernel(const float* __restrict__ Wf, const float* __restrict__ Wih,
                            const float* __restrict__ Whh, const float* __restrict__ Wl,
                            const float* __restrict__ Wo, const float* __restrict__ bih,
                            const float* __restrict__ bhh,
                            _Float16* __restrict__ Wf16, _Float16* __restrict__ Wih16,
                            _Float16* __restrict__ Whh16, _Float16* __restrict__ Wl16,
                            _Float16* __restrict__ Wo16, float* __restrict__ bcomb) {
  const int i = blockIdx.x * blockDim.x + threadIdx.x;
  if (i < Hd * Hd)       Wf16[i]  = (_Float16)Wf[i];
  if (i < 4 * Hd * Fd)   Wih16[i] = (_Float16)Wih[i];
  if (i < 4 * Hd * Hd)   Whh16[i] = (_Float16)Whh[i];
  if (i < Hd * Hd)       Wl16[i]  = (_Float16)Wl[i];
  if (i < Cd * Hd)       Wo16[i]  = (_Float16)Wo[i];
  if (i < 4 * Hd)        bcomb[i] = bih[i] + bhh[i];
}

// ---------------------------------------------------------------------------
// recurrent kernel: 16 independent blocks (one per 16-row batch slice),
// 512 threads = 16 waves; wave w owns hidden n-tile w.
// ---------------------------------------------------------------------------
__global__ __launch_bounds__(512)
void rnn_kernel(const float* __restrict__ x, const float* __restrict__ b_f,
                const _Float16* __restrict__ Wf16, const _Float16* __restrict__ Wih16,
                const _Float16* __restrict__ Whh16, const float* __restrict__ bcomb,
                _Float16* __restrict__ Y16) {
  __shared__ _Float16 zbuf[16 * ZS];   // z / h shadow (fp16), 16 batch rows
  __shared__ _Float16 xbuf[16 * XS];   // x_t staging (fp16)
  __shared__ float    bfl[256];        // b_f
  __shared__ float    bcl[1024];       // b_ih + b_hh

  const int tid  = threadIdx.x;
  const int lane = tid & 31;
  const int w    = tid >> 5;           // 0..15 -> n-tile
  const int rowg = blockIdx.x * 16;    // global batch-row base of this WG
  const int ncol = w * 16;
  const int rr   = (lane >> 4) << 3;   // C-layout row base for lane half
  const int ncl  = lane & 15;          // C-layout column within tile

  for (int i = tid; i < 16 * ZS; i += 512) zbuf[i] = (_Float16)0.0f;
  for (int i = tid; i < 256;  i += 512) bfl[i] = b_f[i];
  for (int i = tid; i < 1024; i += 512) bcl[i] = bcomb[i];
  __syncthreads();

  // cache this wave's W_f column block (16 cols x K=256) in registers: 64 VGPRs
  v16h bwf[KT];
#pragma unroll
  for (int k = 0; k < KT; ++k) bwf[k] = load_b(Wf16, Hd, ncol, k, lane);

  float h[8], c[8], ks[8];
#pragma unroll
  for (int i = 0; i < 8; ++i) { h[i] = 0.0f; c[i] = 0.0f; ks[i] = 0.0f; }

  const float dt = 0.25f;
  const float bias_f = bfl[ncol + ncl];

  for (int t = 0; t < Td; ++t) {
    // prefetch next timestep's x slab for this WG (one cacheline per thread)
    if (t + 1 < Td && tid < 16) {
      const float* xn = x + ((size_t)(t + 1) * Bd + rowg) * Fd + tid * 64;
      __builtin_prefetch(xn, 0, 1);
    }

    // ================= ODE flow: 4 RK4 steps x 4 evals =================
    for (int s = 0; s < 4; ++s) {
#pragma unroll
      for (int e = 0; e < 4; ++e) {
        v16h a[KT];
#pragma unroll
        for (int k = 0; k < KT; ++k) a[k] = load_a(zbuf, ZS, 0, k, lane);
        __syncthreads();  // all z reads done before any z writes

        const float wk = (e == 1 || e == 2) ? 2.0f : 1.0f;
        const float zc = (e == 2) ? dt : 0.5f * dt;

        v8f acc;
#pragma unroll
        for (int r = 0; r < 8; ++r) acc[r] = bias_f;
#pragma unroll
        for (int k = 0; k < KT; ++k) acc = wmma_f16(a[k], bwf[k], acc);

#pragma unroll
        for (int r = 0; r < 8; ++r) {
          const float kv = fast_tanh(acc[r]);
          if (e == 0) ks[r] = kv; else ks[r] += wk * kv;
          float zv;
          if (e < 3) {
            zv = h[r] + zc * kv;
          } else {
            h[r] += (dt / 6.0f) * ks[r];
            zv = h[r];
          }
          zbuf[(rr + r) * ZS + ncol + ncl] = (_Float16)zv;
        }
        __syncthreads();
      }
    }

    // ================= LSTM cell =================
    const float* xt = x + ((size_t)t * Bd + rowg) * Fd;
    for (int i = tid; i < 16 * Fd; i += 512)
      xbuf[(i >> 6) * XS + (i & 63)] = (_Float16)xt[i];
    __syncthreads();

    v16h ah[KT], ax[2];
#pragma unroll
    for (int k = 0; k < KT; ++k) ah[k] = load_a(zbuf, ZS, 0, k, lane);
    ax[0] = load_a(xbuf, XS, 0, 0, lane);
    ax[1] = load_a(xbuf, XS, 0, 1, lane);
    __syncthreads();  // h reads done before h_new writes

    v8f g4[4];
#pragma unroll
    for (int g = 0; g < 4; ++g) {
      const int ng = g * Hd + ncol;
      const float bias = bcl[ng + ncl];
      v8f acc;
#pragma unroll
      for (int r = 0; r < 8; ++r) acc[r] = bias;
#pragma unroll
      for (int k = 0; k < 2; ++k)
        acc = wmma_f16(ax[k], load_b(Wih16, Fd, ng, k, lane), acc);
      // software-pipelined W_hh fragment loads (L2 resident)
      v16h bcur = load_b(Whh16, Hd, ng, 0, lane);
#pragma unroll
      for (int k = 0; k < KT; ++k) {
        v16h bnext;
        if (k + 1 < KT) bnext = load_b(Whh16, Hd, ng, k + 1, lane);
        acc = wmma_f16(ah[k], bcur, acc);
        bcur = bnext;
      }
      g4[g] = acc;
    }
#pragma unroll
    for (int r = 0; r < 8; ++r) {
      const float iv = fast_sigmoid(g4[0][r]);
      const float fv = fast_sigmoid(g4[1][r]);
      const float gv = fast_tanh(g4[2][r]);
      const float ov = fast_sigmoid(g4[3][r]);
      const float cv = fv * c[r] + iv * gv;
      c[r] = cv;
      const float hv = ov * fast_tanh(cv);
      h[r] = hv;
      zbuf[(rr + r) * ZS + ncol + ncl] = (_Float16)hv;
    }
    __syncthreads();

    // async LDS -> global writeback of h_t: exactly one b128 per thread
    _Float16* Yt = Y16 + ((size_t)t * Bd + rowg) * Hd;
    async_st_b128(Yt + w * Hd + lane * 8, zbuf + w * ZS + lane * 8);
    wait_asynccnt0();   // own store issued+done; barrier makes it WG-wide
    __syncthreads();
  }
}

// ---------------------------------------------------------------------------
// readout: out = relu(Y W_l^T + b_l) W_o^T + b_o, fused, fp16 LDS staging.
// ---------------------------------------------------------------------------
__global__ __launch_bounds__(256)
void readout_kernel(const _Float16* __restrict__ Y16, const _Float16* __restrict__ Wl16,
                    const float* __restrict__ b_l, const _Float16* __restrict__ Wo16,
                    const float* __restrict__ b_o, float* __restrict__ out) {
  extern __shared__ char smem[];
  _Float16* tmp = (_Float16*)smem;  // [128][ZS] fp16 relu intermediate

  const int tid  = threadIdx.x;
  const int lane = tid & 31;
  const int w    = tid >> 5;       // 0..7 -> m_tile
  const size_t rowbase = (size_t)blockIdx.x * 128;
  const int rowb = w * 16;
  const int rr   = rowb + ((lane >> 4) << 3);
  const int ncl  = lane & 15;

  const _Float16* Yrows = Y16 + rowbase * Hd;

  // ---- stage 1: relu(Y W_l^T + b_l) -> tmp (fp16) ----
  v16h a[KT];
#pragma unroll
  for (int k = 0; k < KT; ++k) a[k] = load_a(Yrows, Hd, rowb, k, lane);

#pragma unroll
  for (int nt = 0; nt < 16; ++nt) {
    const int ncol = nt * 16;
    const float bias = b_l[ncol + ncl];
    v8f acc;
#pragma unroll
    for (int r = 0; r < 8; ++r) acc[r] = bias;
    v16h bcur = load_b(Wl16, Hd, ncol, 0, lane);
#pragma unroll
    for (int k = 0; k < KT; ++k) {
      v16h bnext;
      if (k + 1 < KT) bnext = load_b(Wl16, Hd, ncol, k + 1, lane);
      acc = wmma_f16(a[k], bcur, acc);
      bcur = bnext;
    }
#pragma unroll
    for (int r = 0; r < 8; ++r)
      tmp[(rr + r) * ZS + ncol + ncl] = (_Float16)fmaxf(acc[r], 0.0f);
  }
  __syncthreads();

  // ---- stage 2: tmp W_o^T + b_o -> out (fp32), C=32 -> 2 n-tiles ----
#pragma unroll
  for (int nt = 0; nt < 2; ++nt) {
    const int ncol = nt * 16;
    const float bias = b_o[ncol + ncl];
    v8f acc;
#pragma unroll
    for (int r = 0; r < 8; ++r) acc[r] = bias;
#pragma unroll
    for (int k = 0; k < KT; ++k) {
      const v16h a2 = load_a(tmp, ZS, rowb, k, lane);
      const v16h b  = load_b(Wo16, Hd, ncol, k, lane);
      acc = wmma_f16(a2, b, acc);
    }
#pragma unroll
    for (int r = 0; r < 8; ++r)
      out[(rowbase + rr + r) * Cd + ncol + ncl] = acc[r];
  }
}

// ---------------------------------------------------------------------------
extern "C" void kernel_launch(void* const* d_in, const int* in_sizes, int n_in,
                              void* d_out, int out_size, void* d_ws, size_t ws_size,
                              hipStream_t stream) {
  const float* x    = (const float*)d_in[0];
  const float* W_f  = (const float*)d_in[1];
  const float* b_f  = (const float*)d_in[2];
  const float* W_ih = (const float*)d_in[3];
  const float* W_hh = (const float*)d_in[4];
  const float* b_ih = (const float*)d_in[5];
  const float* b_hh = (const float*)d_in[6];
  const float* W_l  = (const float*)d_in[7];
  const float* b_l  = (const float*)d_in[8];
  const float* W_o  = (const float*)d_in[9];
  const float* b_o  = (const float*)d_in[10];
  float* out = (float*)d_out;

  char* ws = (char*)d_ws;
  if (ws_size < WS_NEEDED) return;
  _Float16* Wf16  = (_Float16*)(ws + WF16_OFF);
  _Float16* Wih16 = (_Float16*)(ws + WIH16_OFF);
  _Float16* Whh16 = (_Float16*)(ws + WHH16_OFF);
  _Float16* Wl16  = (_Float16*)(ws + WL16_OFF);
  _Float16* Wo16  = (_Float16*)(ws + WO16_OFF);
  float*    bcomb = (float*)(ws + BCOMB_OFF);
  _Float16* Y16   = (_Float16*)(ws + Y16_OFF);

  const int smem_ro = 128 * ZS * 2;
  (void)hipFuncSetAttribute((const void*)readout_kernel,
                            hipFuncAttributeMaxDynamicSharedMemorySize, smem_ro);

  prep_kernel<<<(4 * Hd * Hd + 255) / 256, 256, 0, stream>>>(
      W_f, W_ih, W_hh, W_l, W_o, b_ih, b_hh,
      Wf16, Wih16, Whh16, Wl16, Wo16, bcomb);

  rnn_kernel<<<16, 512, 0, stream>>>(x, b_f, Wf16, Wih16, Whh16, bcomb, Y16);

  readout_kernel<<<(Td * Bd) / 128, 256, smem_ro, stream>>>(
      Y16, Wl16, b_l, Wo16, b_o, out);
}
